// MultiHeadAttention_73598559584357
// MI455X (gfx1250) — matrix operand order repaired and checked by reading, not compiled, not run on hardware
//
#include <hip/hip_runtime.h>

// ---------------------------------------------------------------------------
// MHA forward for MI455X (gfx1250, wave32, WMMA).
//   1) convert weights fp32->bf16
//   2) proj GEMMs q/k/v: X(f32, cvt on the fly) x W^T(bf16, LDS-staged via
//      async load-to-LDS, double-buffered) -> bf16
//        Q,K stored [B,H,L,DK]   (rows contiguous in DK)
//        V  stored [B,H,DK,L]    (transposed: rows contiguous in keys)
//   3) flash attention, 64-key blocks: bf16 WMMA for Q.K^T and P.V,
//      f32 online softmax; V-fragment loads issued before the softmax VALU
//      section so exp/shuffle work hides their latency.
//   4) output proj: O(bf16) x Wo^T(bf16) -> f32 out
// All matrix math via V_WMMA_F32_16X16X32_BF16 (f32 accumulate).
// ---------------------------------------------------------------------------

typedef __attribute__((ext_vector_type(16))) __bf16          v16bf;
typedef __attribute__((ext_vector_type(8)))  float           v8f;
typedef __attribute__((ext_vector_type(8)))  unsigned short  u16x8;
typedef __attribute__((ext_vector_type(4)))  float           f32x4;
typedef int v4i_vs __attribute__((vector_size(16)));  // matches builtin param

#define B_   4
#define L_   2048
#define D_   1024
#define H_   16
#define DK_  64
#define SCALE_ 0.125f   // 1/sqrt(64)

// ---- async load-to-LDS availability (compile-safe probe) -------------------
#if defined(__has_builtin)
#if __has_builtin(__builtin_amdgcn_global_load_async_to_lds_b128)
#define HAVE_ASYNC_COPY 1
#endif
#endif
#ifndef HAVE_ASYNC_COPY
#define HAVE_ASYNC_COPY 0
#endif

__device__ __forceinline__ unsigned short f2bf(float f) {
  union { float f; unsigned int u; } x; x.f = f;
  unsigned int r = x.u + 0x7FFFu + ((x.u >> 16) & 1u);  // round-nearest-even
  return (unsigned short)(r >> 16);
}

// A fragment (16x32, M x K) per ISA 7.12.2: lane m = lane&15, hh = lane>>4
//   elem i in [0,8)  -> K = k0 + hh*8 + i
//   elem i in [8,16) -> K = k0 + 16 + hh*8 + (i-8)
__device__ __forceinline__ v16bf a_frag_bf16(const unsigned short* arow, int k0, int hh) {
  union { v16bf v; u16x8 h[2]; } r;
  r.h[0] = *(const u16x8*)(arow + k0 + hh * 8);
  r.h[1] = *(const u16x8*)(arow + k0 + 16 + hh * 8);
  return r.v;
}

__device__ __forceinline__ v16bf a_frag_f32(const float* arow, int k0, int hh) {
  union { v16bf v; unsigned short u[16]; } r;
  const float* p0 = arow + k0 + hh * 8;
  f32x4 a0 = *(const f32x4*)(p0);
  f32x4 a1 = *(const f32x4*)(p0 + 4);
  f32x4 b0 = *(const f32x4*)(p0 + 16);
  f32x4 b1 = *(const f32x4*)(p0 + 20);
#pragma unroll
  for (int i = 0; i < 4; ++i) {
    r.u[i]      = f2bf(a0[i]);
    r.u[4 + i]  = f2bf(a1[i]);
    r.u[8 + i]  = f2bf(b0[i]);
    r.u[12 + i] = f2bf(b1[i]);
  }
  return r.v;
}

// B fragment (32x16, K x N): lane n = lane&15, hh = lane>>4
//   elem i in [0,8) -> K = k0 + hh*8 + i; elem i in [8,16) -> K = k0 + 16 + hh*8 + (i-8)
__device__ __forceinline__ v16bf b_frag_bf16(const unsigned short* brow, int k0, int hh) {
  union { v16bf v; u16x8 h[2]; } r;
  const unsigned short* p = brow + k0 + hh * 8;
  r.h[0] = *(const u16x8*)(p);
  r.h[1] = *(const u16x8*)(p + 16);
  return r.v;
}

__device__ __forceinline__ v8f wmma_bf16(v16bf a, v16bf b, v8f c) {
  return __builtin_amdgcn_wmma_f32_16x16x32_bf16(false, a, false, b, (short)0, c,
                                                 false, false);
}

__device__ __forceinline__ v8f v8f_zero() {
  v8f z = {0.f, 0.f, 0.f, 0.f, 0.f, 0.f, 0.f, 0.f};
  return z;
}

// Stage one 64(N) x 32(K) bf16 W-tile into LDS. 128 threads, 2 x b128 each.
// Async path is DMA (ASYNCcnt); fallback is load + ds_store (DScnt, drained
// by the barrier fence).
__device__ __forceinline__ void stage_b_tile(unsigned short (*tile)[32],
                                             const unsigned short* __restrict__ W,
                                             int nBase, int k0, int tid) {
#pragma unroll
  for (int i = 0; i < 2; ++i) {
    const int idx = tid + i * 128;       // 0..255
    const int n = idx >> 2;              // 0..63
    const int c = (idx & 3) * 8;         // 0,8,16,24
    const unsigned short* src = W + (size_t)(nBase + n) * D_ + k0 + c;
    unsigned short* dst = &tile[n][c];
#if HAVE_ASYNC_COPY
    __builtin_amdgcn_global_load_async_to_lds_b128(
        (__attribute__((address_space(1))) v4i_vs*)(unsigned long long)(size_t)src,
        (__attribute__((address_space(3))) v4i_vs*)(unsigned)(size_t)dst,
        0, 0);
#else
    *(u16x8*)dst = *(const u16x8*)src;
#endif
  }
}

__device__ __forceinline__ void wait_async_copies() {
#if HAVE_ASYNC_COPY
  asm volatile("s_wait_asynccnt 0" ::: "memory");
#endif
}

// ---------------------------------------------------------------------------
__global__ void cvt_f32_bf16(const float* __restrict__ in,
                             unsigned short* __restrict__ out, int n) {
  int i = blockIdx.x * blockDim.x + threadIdx.x;
  if (i < n) out[i] = f2bf(in[i]);
}

// ---------------------------------------------------------------------------
// GEMM: y[m,n] = sum_k A[m,k] * W[n,k].  M=8192, N=K=1024.
// Block = 128 thr (4 waves), tile 64(M) x 64(N); W-tile double-buffered in
// LDS via async copy; A-fragments double-buffered in registers.
// MODE 0: A=f32, out bf16 [B,H,L,DK]   MODE 1: A=f32, out bf16 [B,H,DK,L]
// MODE 2: A=bf16, out f32 [M,N]
// ---------------------------------------------------------------------------
template <int MODE>
__global__ __launch_bounds__(128) void proj_kernel(
    const void* __restrict__ Aptr, const unsigned short* __restrict__ W,
    void* __restrict__ Out) {
  __shared__ unsigned short btile[2][64][32];  // 8 KB ping-pong

  const int lane = threadIdx.x & 31;
  const int wave = threadIdx.x >> 5;
  const int hh = lane >> 4;
  const int ln = lane & 15;
  const int mBase = blockIdx.x * 64 + wave * 16;
  const int nBase = blockIdx.y * 64;
  const int K = D_;
  const int m = mBase + ln;
  const int NK = K / 32;

  v8f acc[4];
#pragma unroll
  for (int t = 0; t < 4; ++t) acc[t] = v8f_zero();

  // prologue: kick off DMA of first W-tile, load first A-fragment
  stage_b_tile(btile[0], W, nBase, 0, threadIdx.x);
  v16bf af;
  if constexpr (MODE == 2)
    af = a_frag_bf16((const unsigned short*)Aptr + (size_t)m * K, 0, hh);
  else
    af = a_frag_f32((const float*)Aptr + (size_t)m * K, 0, hh);

  for (int kc = 0; kc < NK; ++kc) {
    const int buf = kc & 1;
    wait_async_copies();   // this wave's fill of btile[buf] complete
    __syncthreads();       // all waves' fills visible; prior readers done

    if (kc + 1 < NK)       // overlap next tile DMA with this tile's WMMAs
      stage_b_tile(btile[buf ^ 1], W, nBase, (kc + 1) * 32, threadIdx.x);

    v16bf afn = af;        // register double-buffer for A
    if (kc + 1 < NK) {
      if constexpr (MODE == 2)
        afn = a_frag_bf16((const unsigned short*)Aptr + (size_t)m * K,
                          (kc + 1) * 32, hh);
      else
        afn = a_frag_f32((const float*)Aptr + (size_t)m * K, (kc + 1) * 32, hh);
    }

#pragma unroll
    for (int t = 0; t < 4; ++t) {
      v16bf bf = b_frag_bf16(&btile[buf][t * 16 + ln][0], 0, hh);
      acc[t] = wmma_bf16(af, bf, acc[t]);
    }
    af = afn;
  }

#pragma unroll
  for (int t = 0; t < 4; ++t) {
#pragma unroll
    for (int j = 0; j < 8; ++j) {
      const int mm = mBase + j + 8 * hh;  // C layout: VGPR j -> row j + 8*hh
      const int nn = nBase + t * 16 + ln;
      const float v = acc[t][j];
      if constexpr (MODE == 2) {
        ((float*)Out)[(size_t)mm * D_ + nn] = v;
      } else {
        const int b = mm >> 11, l = mm & (L_ - 1);
        const int h = nn >> 6, dk = nn & (DK_ - 1);
        unsigned short* O = (unsigned short*)Out;
        if constexpr (MODE == 0)
          O[(((size_t)(b * H_ + h)) * L_ + l) * DK_ + dk] = f2bf(v);
        else
          O[(((size_t)(b * H_ + h)) * DK_ + dk) * L_ + l] = f2bf(v);
      }
    }
  }
}

// ---------------------------------------------------------------------------
// Flash attention, 64-key blocks. Grid: B*H*(L/64) blocks of 128 thr (4 waves).
// Each wave: 16 query rows x full DK=64 (4 f32 accumulators).
// ---------------------------------------------------------------------------
__global__ __launch_bounds__(128) void attn_kernel(
    const unsigned short* __restrict__ Qh, const unsigned short* __restrict__ Kh,
    const unsigned short* __restrict__ VhT, unsigned short* __restrict__ O) {
  __shared__ unsigned short p_lds[4][16][64];  // per-wave P tiles (8 KB)

  const int lane = threadIdx.x & 31;
  const int wave = threadIdx.x >> 5;
  const int hh = lane >> 4;
  const int ln = lane & 15;

  int bid = blockIdx.x;
  const int nQB = L_ / 64;
  const int qb = bid % nQB; bid /= nQB;
  const int h = bid % H_;   bid /= H_;
  const int b = bid;
  const int qBase = qb * 64 + wave * 16;

  const unsigned short* qbase = Qh + ((size_t)(b * H_ + h)) * L_ * DK_;
  const unsigned short* kbase = Kh + ((size_t)(b * H_ + h)) * L_ * DK_;
  const unsigned short* vbase = VhT + ((size_t)(b * H_ + h)) * DK_ * L_;

  const unsigned short* qrow = qbase + (size_t)(qBase + ln) * DK_;
  const v16bf qf0 = a_frag_bf16(qrow, 0, hh);
  const v16bf qf1 = a_frag_bf16(qrow, 32, hh);

  v8f acc[4];
#pragma unroll
  for (int t = 0; t < 4; ++t) acc[t] = v8f_zero();
  float mrow[8], lsum[8];
#pragma unroll
  for (int j = 0; j < 8; ++j) { mrow[j] = -1e30f; lsum[j] = 0.f; }

  for (int kb = 0; kb < L_; kb += 64) {
    if (kb + 64 < L_) {
      __builtin_prefetch(kbase + (size_t)(kb + 64 + lane) * DK_, 0, 1);
      __builtin_prefetch(vbase + (size_t)lane * L_ + kb + 64, 0, 1);
    }

    // --- all K fragments for this block first (one wait, not per-WMMA) ---
    v16bf kf[4][2];
#pragma unroll
    for (int nt = 0; nt < 4; ++nt) {
      const unsigned short* krow = kbase + (size_t)(kb + nt * 16 + ln) * DK_;
      kf[nt][0] = b_frag_bf16(krow, 0, hh);
      kf[nt][1] = b_frag_bf16(krow, 32, hh);
    }

    // --- scores: 4 key subtiles x K-depth 64 = 8 WMMAs ---
    v8f s[4];
#pragma unroll
    for (int nt = 0; nt < 4; ++nt) {
      v8f z = v8f_zero();
      z = wmma_bf16(qf0, kf[nt][0], z);
      z = wmma_bf16(qf1, kf[nt][1], z);
      s[nt] = z;
    }

    // --- issue V fragment loads now; softmax VALU below hides their latency
    v16bf vf[4][2];
#pragma unroll
    for (int t = 0; t < 4; ++t) {
      const unsigned short* vrow = vbase + (size_t)(t * 16 + ln) * L_ + kb;
      vf[t][0] = b_frag_bf16(vrow, 0, hh);
      vf[t][1] = b_frag_bf16(vrow, 32, hh);
    }

    // --- online softmax per row (row j+8*hh in VGPR j, lanes = keys) ---
#pragma unroll
    for (int j = 0; j < 8; ++j) {
      float v0 = s[0][j] * SCALE_;
      float v1 = s[1][j] * SCALE_;
      float v2 = s[2][j] * SCALE_;
      float v3 = s[3][j] * SCALE_;
      float mx = fmaxf(fmaxf(v0, v1), fmaxf(v2, v3));
      mx = fmaxf(mx, __shfl_xor(mx, 1, 32));
      mx = fmaxf(mx, __shfl_xor(mx, 2, 32));
      mx = fmaxf(mx, __shfl_xor(mx, 4, 32));
      mx = fmaxf(mx, __shfl_xor(mx, 8, 32));
      const float mnew = fmaxf(mrow[j], mx);
      const float alpha = __expf(mrow[j] - mnew);
      const float p0 = __expf(v0 - mnew);
      const float p1 = __expf(v1 - mnew);
      const float p2 = __expf(v2 - mnew);
      const float p3 = __expf(v3 - mnew);
      float rs = (p0 + p1) + (p2 + p3);
      rs += __shfl_xor(rs, 1, 32);
      rs += __shfl_xor(rs, 2, 32);
      rs += __shfl_xor(rs, 4, 32);
      rs += __shfl_xor(rs, 8, 32);
      lsum[j] = lsum[j] * alpha + rs;
      mrow[j] = mnew;
#pragma unroll
      for (int t = 0; t < 4; ++t) acc[t][j] *= alpha;
      const int mI = j + 8 * hh;
      p_lds[wave][mI][ln]      = f2bf(p0);
      p_lds[wave][mI][16 + ln] = f2bf(p1);
      p_lds[wave][mI][32 + ln] = f2bf(p2);
      p_lds[wave][mI][48 + ln] = f2bf(p3);
    }

    asm volatile("s_wait_dscnt 0" ::: "memory");  // P visible to this wave

    // --- O += P * V : 8 WMMAs (2 key chunks x 4 dim subtiles) ---
    const v16bf pf0 = a_frag_bf16(&p_lds[wave][ln][0], 0, hh);
    const v16bf pf1 = a_frag_bf16(&p_lds[wave][ln][0], 32, hh);
#pragma unroll
    for (int t = 0; t < 4; ++t) {
      acc[t] = wmma_bf16(pf0, vf[t][0], acc[t]);
      acc[t] = wmma_bf16(pf1, vf[t][1], acc[t]);
    }
  }

  // epilogue: normalize, merge heads -> bf16 [B, L, D]
#pragma unroll
  for (int j = 0; j < 8; ++j) {
    const float inv = 1.0f / lsum[j];
    const int q = qBase + j + 8 * hh;
    const size_t rowOff = ((size_t)(b * L_ + q)) * D_ + h * DK_;
#pragma unroll
    for (int t = 0; t < 4; ++t)
      O[rowOff + t * 16 + ln] = f2bf(acc[t][j] * inv);
  }
}

// ---------------------------------------------------------------------------
extern "C" void kernel_launch(void* const* d_in, const int* in_sizes, int n_in,
                              void* d_out, int out_size, void* d_ws, size_t ws_size,
                              hipStream_t stream) {
  (void)in_sizes; (void)n_in; (void)out_size; (void)ws_size;
  const float* q  = (const float*)d_in[0];
  const float* k  = (const float*)d_in[1];
  const float* v  = (const float*)d_in[2];
  const float* wq = (const float*)d_in[3];
  const float* wk = (const float*)d_in[4];
  const float* wv = (const float*)d_in[5];
  const float* wo = (const float*)d_in[6];

  char* ws = (char*)d_ws;
  const size_t WBYTES = (size_t)D_ * D_ * 2;       // 2 MB per weight
  const size_t TBYTES = (size_t)B_ * L_ * D_ * 2;  // 16 MB per tensor
  unsigned short* wqb = (unsigned short*)(ws);
  unsigned short* wkb = (unsigned short*)(ws + WBYTES);
  unsigned short* wvb = (unsigned short*)(ws + 2 * WBYTES);
  unsigned short* wob = (unsigned short*)(ws + 3 * WBYTES);
  unsigned short* qh  = (unsigned short*)(ws + 4 * WBYTES);
  unsigned short* kh  = (unsigned short*)(ws + 4 * WBYTES + TBYTES);
  unsigned short* vhT = (unsigned short*)(ws + 4 * WBYTES + 2 * TBYTES);
  unsigned short* ob  = (unsigned short*)(ws + 4 * WBYTES + 3 * TBYTES);

  const int nw = D_ * D_;
  cvt_f32_bf16<<<nw / 256, 256, 0, stream>>>(wq, wqb, nw);
  cvt_f32_bf16<<<nw / 256, 256, 0, stream>>>(wk, wkb, nw);
  cvt_f32_bf16<<<nw / 256, 256, 0, stream>>>(wv, wvb, nw);
  cvt_f32_bf16<<<nw / 256, 256, 0, stream>>>(wo, wob, nw);

  dim3 gp((B_ * L_) / 64, D_ / 64), bp(128);
  proj_kernel<0><<<gp, bp, 0, stream>>>(q, wqb, qh);
  proj_kernel<0><<<gp, bp, 0, stream>>>(k, wkb, kh);
  proj_kernel<1><<<gp, bp, 0, stream>>>(v, wvb, vhT);

  attn_kernel<<<B_ * H_ * (L_ / 64), 128, 0, stream>>>(qh, kh, vhT, ob);

  proj_kernel<2><<<gp, bp, 0, stream>>>(ob, wob, d_out);
}
